// NeuralTransitionParser_29875792511184
// MI455X (gfx1250) — compile-verified, matrix operand order checked by reading, not run on hardware
//
#include <hip/hip_runtime.h>

// ---------------------------------------------------------------------------
// Stack-LSTM transition parser forward pass for MI455X (gfx1250, wave32).
//
//  * Weights pre-packed once into bf16 WMMA B-fragment order
//    ([K/32][G/16][lane][16]) -> every fragment is 2x global_load_b128.
//  * Hidden states kept in f32 (master, feeds c-path + loss MLP) plus a bf16
//    shadow written by the producing cell -> A-fragments are 2x b128 bf16
//    loads with zero per-consumer converts.
//  * x-part of all buffer-word cells precomputed as one big WMMA GEMM.
//  * One fused kernel per transition step: 64 stack-cell waves + 64
//    buffer-cell waves + 64 action-cell waves + 1 loss block (disjoint
//    read/write sets), minimizing launches on the ~500-deep dependent chain.
// ---------------------------------------------------------------------------

typedef __bf16 bhalf;
typedef __attribute__((ext_vector_type(16))) __bf16 v16bf;
typedef __attribute__((ext_vector_type(8)))  __bf16 v8bf;
typedef __attribute__((ext_vector_type(8)))  float  v8f;

#define Bsz   16
#define Lsz   64
#define Tsz   126          // 2*L-2
#define Dsz   134          // 2*L+6
#define H2    1024
#define G4    4096         // 4*H2
#define NT4   256          // G4/16
#define SSTR  (134LL*1024LL)   // per-sentence stack stride (elements)
#define AHSLOT 16384LL         // 16*1024 (one a_h slot, elements)

// ---------------------------- device helpers -------------------------------

__device__ __forceinline__ float sigm_f(float x) {
  return 1.0f / (1.0f + __expf(-x));
}
__device__ __forceinline__ float tanh_f(float x) {
  return 2.0f / (1.0f + __expf(-2.0f * x)) - 1.0f;   // 2*sigmoid(2x)-1
}

// A-fragment (16x32 bf16) from one bf16 row per lane: two contiguous 8-elem
// runs (16B each) -> 2x global_load_b128, no conversions.
__device__ __forceinline__ v16bf load_a_row_bf(const bhalf* __restrict__ row,
                                               int k0, int koff) {
  v8bf lo = *(const v8bf*)(row + k0 + koff);        // i = 0..7
  v8bf hi = *(const v8bf*)(row + k0 + koff + 16);   // i = 8..15
  v16bf a;
#pragma unroll
  for (int i = 0; i < 8; ++i) { a[i] = lo[i]; a[8 + i] = hi[i]; }
  return a;
}

// ---------------------------- pack kernels ---------------------------------

// Pack W [G rows x K cols] (f32, row-major) into WMMA B-fragment order:
// out[((kt*(G/16) + nt)*32 + lane)*16 + i] = bf16( W[g*K + k] )
__global__ void pack_B_k(const float* __restrict__ W, bhalf* __restrict__ out,
                         int G, int K) {
  long long total = (long long)G * K;
  int NT = G / 16;
  long long stride = (long long)gridDim.x * blockDim.x;
  for (long long idx = (long long)blockIdx.x * blockDim.x + threadIdx.x;
       idx < total; idx += stride) {
    int i    = (int)(idx & 15);
    int lane = (int)((idx >> 4) & 31);
    long long t2 = idx >> 9;
    int nt = (int)(t2 % NT);
    int kt = (int)(t2 / NT);
    int koff = (lane >> 4) * 8;
    int k = kt * 32 + koff + i + ((i >= 8) ? 8 : 0);
    int g = nt * 16 + (lane & 15);
    out[idx] = (__bf16)W[(long long)g * K + k];
  }
}

// Gather word/tag embeddings into A-fragment packed x_emb:
// out layout ((mt*32 + kt)*32 + lane)*16 + i  with M=1024, K=1024
__global__ void pack_x_emb_k(const int* __restrict__ word_ids,
                             const int* __restrict__ tag_ids,
                             const float* __restrict__ W_word,
                             const float* __restrict__ W_tag,
                             bhalf* __restrict__ out) {
  long long total = 1024LL * 1024LL;
  long long stride = (long long)gridDim.x * blockDim.x;
  for (long long idx = (long long)blockIdx.x * blockDim.x + threadIdx.x;
       idx < total; idx += stride) {
    int i    = (int)(idx & 15);
    int lane = (int)((idx >> 4) & 31);
    long long t2 = idx >> 9;
    int kt = (int)(t2 & 31);
    int mt = (int)(t2 >> 5);
    int koff = (lane >> 4) * 8;
    int k = kt * 32 + koff + i + ((i >= 8) ? 8 : 0);
    int m = mt * 16 + (lane & 15);
    int b = m >> 6, l = m & 63;
    float v = (k < 512)
                  ? W_word[(long long)word_ids[b * Lsz + l] * 512 + k]
                  : W_tag[(long long)tag_ids[b * Lsz + l] * 512 + (k - 512)];
    out[idx] = (__bf16)v;
  }
}

__global__ void vadd_k(const float* __restrict__ a, const float* __restrict__ b,
                       float* __restrict__ o, int n) {
  int i = blockIdx.x * blockDim.x + threadIdx.x;
  if (i < n) o[i] = a[i] + b[i];
}

__global__ void cvt_bf_k(const float* __restrict__ in, bhalf* __restrict__ out,
                         int n) {
  int i = blockIdx.x * blockDim.x + threadIdx.x;
  if (i < n) out[i] = (__bf16)in[i];
}

__global__ void init_state_k(const float* __restrict__ init_h,
                             const float* __restrict__ init_c,
                             float* __restrict__ hs_s, float* __restrict__ cs_s,
                             float* __restrict__ hs_b, float* __restrict__ cs_b,
                             bhalf* __restrict__ hsb_s, bhalf* __restrict__ hsb_b,
                             float* __restrict__ zerovec,
                             bhalf* __restrict__ zerovec_bf) {
  int i = blockIdx.x * blockDim.x + threadIdx.x;
  if (i < H2) {
    float h = init_h[i], c = init_c[i];
#pragma unroll 1
    for (int b = 0; b < Bsz; ++b) {
      hs_s[(long long)b * SSTR + i] = h;
      cs_s[(long long)b * SSTR + i] = c;
      hs_b[(long long)b * SSTR + i] = h;
      cs_b[(long long)b * SSTR + i] = c;
      hsb_s[(long long)b * SSTR + i] = (__bf16)h;
      hsb_b[(long long)b * SSTR + i] = (__bf16)h;
    }
    zerovec[i] = 0.0f;
    zerovec_bf[i] = (__bf16)0.0f;
  }
}

// Precompute stack/buffer pointer trajectories from the oracle actions.
__global__ void traj_k(const int* __restrict__ actions, int* __restrict__ sI,
                       int* __restrict__ bI) {
  int b = threadIdx.x;
  if (b >= Bsz) return;
  int sp = 2, bp = 64;
  for (int t = 0; t < Tsz; ++t) {
    sI[t * Bsz + b] = sp;
    bI[t * Bsz + b] = bp;
    int a = actions[b * Tsz + t];
    if (a == 0) { sp += 1; bp -= 1; }
    else        { sp -= 1; if (a == 2) bp += 1; }
  }
}

// ---------------------- big precompute GEMM (WMMA) -------------------------
// Xb_pre[m][g] = x_emb[m] . Wih_b[g] + (bih_b+bhh_b)[g]   for all 1024 words.
__global__ void __launch_bounds__(32)
gemm_xpre_k(const bhalf* __restrict__ Apack, const bhalf* __restrict__ Bpack,
            const float* __restrict__ bias, float* __restrict__ out) {
  int mt = blockIdx.x;   // 0..63
  int nt = blockIdx.y;   // 0..255
  int lane = threadIdx.x;
  const v16bf* Av = (const v16bf*)Apack;
  const v16bf* Bv = (const v16bf*)Bpack;
  v8f acc = {};
#pragma unroll 4
  for (int kt = 0; kt < 32; ++kt) {
    v16bf a = Av[(mt * 32 + kt) * 32 + lane];
    v16bf b = Bv[(kt * NT4 + nt) * 32 + lane];
    acc = __builtin_amdgcn_wmma_f32_16x16x32_bf16(false, a, false, b,
                                                  (short)0, acc, false, false);
  }
  int col = nt * 16 + (lane & 15);
  int Mb  = mt * 16 + ((lane >> 4) * 8);
  float bb = bias[col];
#pragma unroll
  for (int i = 0; i < 8; ++i)
    out[(long long)(Mb + i) * G4 + col] = acc[i] + bb;
}

// ------------------------- LSTM cell (WMMA core) ---------------------------
// One wave per 16 hidden columns; 4 gate tiles (i,f,g,o) accumulated in f32.
// Resolver contract:
//   xrow(b)/hrow(b): bf16 A-source rows;  crow(b): f32 c state
//   xprow(b): f32 precomputed x-part row (use_xpre only)
//   dhrow/dcrow: f32 dests; dhrow_bf: bf16 shadow dest; active(b): write mask
template <typename R>
__device__ __forceinline__ void cell_core(const R& r, int ntile,
                                          const bhalf* __restrict__ WihP,
                                          int KxTiles,
                                          const bhalf* __restrict__ WhhP,
                                          const float* __restrict__ bias,
                                          int use_xpre) {
  const int lane = threadIdx.x;
  const int koff = (lane >> 4) * 8;
  const v16bf* Whhv = (const v16bf*)WhhP;
  v8f acc0 = {}, acc1 = {}, acc2 = {}, acc3 = {};

  // Warm the state rows (weights are L2-resident; state is cold per step).
  {
    const bhalf* hrp = r.hrow(lane & 15);
    __builtin_prefetch(hrp, 0, 3);
    __builtin_prefetch(hrp + 512, 0, 3);
  }

  if (!use_xpre) {                        // x @ Wih^T phase
    const v16bf* Wihv = (const v16bf*)WihP;
    const bhalf* xr = r.xrow(lane & 15);
    __builtin_prefetch(xr, 0, 3);
    for (int kt = 0; kt < KxTiles; ++kt) {
      v16bf a = load_a_row_bf(xr, kt * 32, koff);
      v16bf b0 = Wihv[(kt * NT4 + (0 * 64 + ntile)) * 32 + lane];
      v16bf b1 = Wihv[(kt * NT4 + (1 * 64 + ntile)) * 32 + lane];
      v16bf b2 = Wihv[(kt * NT4 + (2 * 64 + ntile)) * 32 + lane];
      v16bf b3 = Wihv[(kt * NT4 + (3 * 64 + ntile)) * 32 + lane];
      acc0 = __builtin_amdgcn_wmma_f32_16x16x32_bf16(false, a, false, b0, (short)0, acc0, false, false);
      acc1 = __builtin_amdgcn_wmma_f32_16x16x32_bf16(false, a, false, b1, (short)0, acc1, false, false);
      acc2 = __builtin_amdgcn_wmma_f32_16x16x32_bf16(false, a, false, b2, (short)0, acc2, false, false);
      acc3 = __builtin_amdgcn_wmma_f32_16x16x32_bf16(false, a, false, b3, (short)0, acc3, false, false);
    }
  }
  {                                       // h @ Whh^T phase (K = 1024)
    const bhalf* hr = r.hrow(lane & 15);
    for (int kt = 0; kt < 32; ++kt) {
      v16bf a = load_a_row_bf(hr, kt * 32, koff);
      v16bf b0 = Whhv[(kt * NT4 + (0 * 64 + ntile)) * 32 + lane];
      v16bf b1 = Whhv[(kt * NT4 + (1 * 64 + ntile)) * 32 + lane];
      v16bf b2 = Whhv[(kt * NT4 + (2 * 64 + ntile)) * 32 + lane];
      v16bf b3 = Whhv[(kt * NT4 + (3 * 64 + ntile)) * 32 + lane];
      acc0 = __builtin_amdgcn_wmma_f32_16x16x32_bf16(false, a, false, b0, (short)0, acc0, false, false);
      acc1 = __builtin_amdgcn_wmma_f32_16x16x32_bf16(false, a, false, b1, (short)0, acc1, false, false);
      acc2 = __builtin_amdgcn_wmma_f32_16x16x32_bf16(false, a, false, b2, (short)0, acc2, false, false);
      acc3 = __builtin_amdgcn_wmma_f32_16x16x32_bf16(false, a, false, b3, (short)0, acc3, false, false);
    }
  }
  // Epilogue: gates -> (h2, c2), masked scatter to destination rows.
  const int col = ntile * 16 + (lane & 15);  // hidden unit 0..1023
  const int Mb  = (lane >> 4) * 8;           // first sentence this lane owns
  float bi = 0.f, bf_ = 0.f, bg = 0.f, bo = 0.f;
  if (bias) {
    bi = bias[col]; bf_ = bias[col + 1024];
    bg = bias[col + 2048]; bo = bias[col + 3072];
  }
#pragma unroll
  for (int i = 0; i < 8; ++i) {
    int b = Mb + i;
    float zi = acc0[i] + bi, zf = acc1[i] + bf_;
    float zg = acc2[i] + bg, zo = acc3[i] + bo;
    if (use_xpre) {
      const float* xp = r.xprow(b);
      zi += xp[col]; zf += xp[col + 1024];
      zg += xp[col + 2048]; zo += xp[col + 3072];
    }
    float cold = r.crow(b)[col];
    float c2 = sigm_f(zf) * cold + sigm_f(zi) * tanh_f(zg);
    float h2 = sigm_f(zo) * tanh_f(c2);
    if (r.active(b)) {
      r.dhrow(b)[col] = h2;
      r.dcrow(b)[col] = c2;
      r.dhrow_bf(b)[col] = (__bf16)h2;
    }
  }
}

template <typename R>
__global__ void __launch_bounds__(32)
cell_k(R r, const bhalf* __restrict__ WihP, int KxTiles,
       const bhalf* __restrict__ WhhP, const float* __restrict__ bias,
       int use_xpre) {
  cell_core(r, blockIdx.x, WihP, KxTiles, WhhP, bias, use_xpre);
}

// ------------------------------ resolvers ----------------------------------

struct UniformR {                 // guard pushes, w0 push, initial action cell
  const bhalf* xbase; long long xbstride;
  const bhalf* hbb; const float* cb; long long hstride; int src_p;
  float* dhb; float* dcb; bhalf* dhbb; long long dstride; int dst_p;
  __device__ const bhalf* xrow(int b) const { return xbase + (long long)b * xbstride; }
  __device__ const float* xprow(int b) const { return (const float*)0; }
  __device__ const bhalf* hrow(int b) const { return hbb + (long long)b * hstride + (long long)src_p * H2; }
  __device__ const float* crow(int b) const { return cb + (long long)b * hstride + (long long)src_p * H2; }
  __device__ float* dhrow(int b) const { return dhb + (long long)b * dstride + (long long)dst_p * H2; }
  __device__ float* dcrow(int b) const { return dcb + (long long)b * dstride + (long long)dst_p * H2; }
  __device__ bhalf* dhrow_bf(int b) const { return dhbb + (long long)b * dstride + (long long)dst_p * H2; }
  __device__ bool active(int) const { return true; }
};

struct XpreR {                    // buffer word pushes (x-part precomputed)
  const float* xpre; long long xpstride;
  const bhalf* hbb; const float* cb;
  float* dhb; float* dcb; bhalf* dhbb;
  int src_p; int dst_p;
  __device__ const bhalf* xrow(int b) const { return hbb; }   // unused
  __device__ const float* xprow(int b) const { return xpre + (long long)b * xpstride; }
  __device__ const bhalf* hrow(int b) const { return hbb + (long long)b * SSTR + (long long)src_p * H2; }
  __device__ const float* crow(int b) const { return cb + (long long)b * SSTR + (long long)src_p * H2; }
  __device__ float* dhrow(int b) const { return dhb + (long long)b * SSTR + (long long)dst_p * H2; }
  __device__ float* dcrow(int b) const { return dcb + (long long)b * SSTR + (long long)dst_p * H2; }
  __device__ bhalf* dhrow_bf(int b) const { return dhbb + (long long)b * SSTR + (long long)dst_p * H2; }
  __device__ bool active(int) const { return true; }
};

struct TransStackR {              // shift: push buffer-top onto stack (a==0)
  const int* acts; const int* sI; const int* bI; int t;
  const bhalf* hsb_b;                         // x source (buffer tops, bf16)
  const bhalf* hsb_s; const float* cb;        // stack h (bf16) / c (f32)
  float* dhb; float* dcb; bhalf* dhbb;
  __device__ int A(int b) const { return acts[b * Tsz + t]; }
  __device__ const bhalf* xrow(int b) const { return hsb_b + (long long)b * SSTR + (long long)bI[t * Bsz + b] * H2; }
  __device__ const float* xprow(int b) const { return (const float*)0; }
  __device__ const bhalf* hrow(int b) const { return hsb_s + (long long)b * SSTR + (long long)sI[t * Bsz + b] * H2; }
  __device__ const float* crow(int b) const { return cb + (long long)b * SSTR + (long long)sI[t * Bsz + b] * H2; }
  __device__ float* dhrow(int b) const { return dhb + (long long)b * SSTR + (long long)(sI[t * Bsz + b] + 1) * H2; }
  __device__ float* dcrow(int b) const { return dcb + (long long)b * SSTR + (long long)(sI[t * Bsz + b] + 1) * H2; }
  __device__ bhalf* dhrow_bf(int b) const { return dhbb + (long long)b * SSTR + (long long)(sI[t * Bsz + b] + 1) * H2; }
  __device__ bool active(int b) const { return A(b) == 0; }
};

struct TransBufR {                // reduce_r: push stack-top onto buffer (a==2)
  const int* acts; const int* sI; const int* bI; int t;
  const bhalf* hsb_s;                         // x source (stack tops, bf16)
  const bhalf* hsb_b; const float* cb;        // buffer h (bf16) / c (f32)
  float* dhb; float* dcb; bhalf* dhbb;
  __device__ int A(int b) const { return acts[b * Tsz + t]; }
  __device__ const bhalf* xrow(int b) const { return hsb_s + (long long)b * SSTR + (long long)sI[t * Bsz + b] * H2; }
  __device__ const float* xprow(int b) const { return (const float*)0; }
  __device__ const bhalf* hrow(int b) const { return hsb_b + (long long)b * SSTR + (long long)bI[t * Bsz + b] * H2; }
  __device__ const float* crow(int b) const { return cb + (long long)b * SSTR + (long long)bI[t * Bsz + b] * H2; }
  __device__ float* dhrow(int b) const { return dhb + (long long)b * SSTR + (long long)(bI[t * Bsz + b] + 1) * H2; }
  __device__ float* dcrow(int b) const { return dcb + (long long)b * SSTR + (long long)(bI[t * Bsz + b] + 1) * H2; }
  __device__ bhalf* dhrow_bf(int b) const { return dhbb + (long long)b * SSTR + (long long)(bI[t * Bsz + b] + 1) * H2; }
  __device__ bool active(int b) const { return A(b) == 2; }
};

struct TransActR {                // action LSTM (always fires; ping-pong slots)
  const int* acts; int t;
  const bhalf* aembb;             // A_emb bf16 [3][512]
  const bhalf* ahbb; const float* acb;
  float* dahb; float* dacb; bhalf* dahbb;
  __device__ int A(int b) const { return acts[b * Tsz + t]; }
  __device__ const bhalf* xrow(int b) const { return aembb + (long long)A(b) * 512; }
  __device__ const float* xprow(int b) const { return (const float*)0; }
  __device__ const bhalf* hrow(int b) const { return ahbb + (long long)(t & 1) * AHSLOT + (long long)b * H2; }
  __device__ const float* crow(int b) const { return acb + (long long)(t & 1) * AHSLOT + (long long)b * H2; }
  __device__ float* dhrow(int b) const { return dahb + (long long)((t + 1) & 1) * AHSLOT + (long long)b * H2; }
  __device__ float* dcrow(int b) const { return dacb + (long long)((t + 1) & 1) * AHSLOT + (long long)b * H2; }
  __device__ bhalf* dhrow_bf(int b) const { return dahbb + (long long)((t + 1) & 1) * AHSLOT + (long long)b * H2; }
  __device__ bool active(int) const { return true; }
};

// ------------------------------ loss (device) ------------------------------

__device__ __forceinline__ float dot1024(const float* __restrict__ w,
                                         const float* __restrict__ x) {
  const float4* wv = (const float4*)w;
  const float4* xv = (const float4*)x;
  float s = 0.f;
  for (int k = 0; k < 256; ++k) {
    float4 a = wv[k], b = xv[k];
    s = fmaf(a.x, b.x, s); s = fmaf(a.y, b.y, s);
    s = fmaf(a.z, b.z, s); s = fmaf(a.w, b.w, s);
  }
  return s;
}

__device__ void loss_body(int t, const float* __restrict__ hs_s,
                          const float* __restrict__ hs_b,
                          const float* __restrict__ a_h,
                          const int* __restrict__ sI, const int* __restrict__ bI,
                          const int* __restrict__ actions,
                          const float* __restrict__ W_mlp,
                          const float* __restrict__ b_mlp,
                          float* __restrict__ losses) {
  __shared__ float dots[48];
  int j = threadIdx.x;
  for (int d = j; d < 48; d += 32) {
    int b = d / 3, r = d % 3;
    const float* st = hs_s + (long long)b * SSTR + (long long)sI[t * Bsz + b] * H2;
    const float* bt = hs_b + (long long)b * SSTR + (long long)bI[t * Bsz + b] * H2;
    const float* ah = a_h + (long long)(t & 1) * AHSLOT + (long long)b * H2;
    const float* w = W_mlp + (long long)r * 3072;
    float s = dot1024(w, st) + dot1024(w + 1024, bt) + dot1024(w + 2048, ah);
    dots[d] = s + b_mlp[r];
  }
  __syncthreads();
  if (j < Bsz) {
    int b = j;
    float l0 = fmaxf(dots[b * 3 + 0], 0.f);
    float l1 = fmaxf(dots[b * 3 + 1], 0.f);
    float l2 = fmaxf(dots[b * 3 + 2], 0.f);
    float m = fmaxf(l0, fmaxf(l1, l2));
    float e0 = __expf(l0 - m), e1 = __expf(l1 - m), e2 = __expf(l2 - m);
    float inv = 1.0f / (e0 + e1 + e2);
    float q0 = e0 * inv, q1 = e1 * inv, q2 = e2 * inv;   // first softmax
    int a = actions[b * Tsz + t];
    float qa = (a == 0) ? q0 : ((a == 1) ? q1 : q2);
    float qm = fmaxf(q0, fmaxf(q1, q2));
    float se = __expf(q0 - qm) + __expf(q1 - qm) + __expf(q2 - qm);
    losses[b * Tsz + t] = (qm + __logf(se)) - qa;         // -log_softmax(q)[a]
  }
}

// ------------------- fused transition step (1 launch/step) -----------------
// Grid: 193 blocks x 32 threads.
//  blocks [0,64)    : stack cell   (active per-sentence iff a==0)
//  blocks [64,128)  : buffer cell  (active per-sentence iff a==2)
//  blocks [128,192) : action cell  (always)
//  block  192       : MLP + double-softmax loss (reads pre-step tops)
// All read/write sets are disjoint within a step.
__global__ void __launch_bounds__(32)
trans_step_k(int t, const int* __restrict__ actions,
             const int* __restrict__ sI, const int* __restrict__ bI,
             const bhalf* __restrict__ aemb_bf,
             const bhalf* __restrict__ P_Wih_s, const bhalf* __restrict__ P_Whh_s,
             const float* __restrict__ bias_s,
             const bhalf* __restrict__ P_Wih_b, const bhalf* __restrict__ P_Whh_b,
             const float* __restrict__ bias_b,
             const bhalf* __restrict__ P_Wih_a, const bhalf* __restrict__ P_Whh_a,
             const float* __restrict__ bias_a,
             float* __restrict__ hs_s, float* __restrict__ cs_s,
             float* __restrict__ hs_b, float* __restrict__ cs_b,
             bhalf* __restrict__ hsb_s, bhalf* __restrict__ hsb_b,
             float* __restrict__ a_h, float* __restrict__ a_c,
             bhalf* __restrict__ a_hb,
             const float* __restrict__ W_mlp, const float* __restrict__ b_mlp,
             float* __restrict__ losses) {
  int role  = blockIdx.x >> 6;
  int ntile = blockIdx.x & 63;
  if (role == 0) {
    TransStackR r{actions, sI, bI, t, hsb_b, hsb_s, cs_s, hs_s, cs_s, hsb_s};
    cell_core(r, ntile, P_Wih_s, 32, P_Whh_s, bias_s, 0);
  } else if (role == 1) {
    TransBufR r{actions, sI, bI, t, hsb_s, hsb_b, cs_b, hs_b, cs_b, hsb_b};
    cell_core(r, ntile, P_Wih_b, 32, P_Whh_b, bias_b, 0);
  } else if (role == 2) {
    TransActR r{actions, t, aemb_bf, a_hb, a_c, a_h, a_c, a_hb};
    cell_core(r, ntile, P_Wih_a, 16, P_Whh_a, bias_a, 0);
  } else {
    loss_body(t, hs_s, hs_b, a_h, sI, bI, actions, W_mlp, b_mlp, losses);
  }
}

__global__ void reduce_k(const float* __restrict__ losses, float* __restrict__ out) {
  __shared__ float s[256];
  int tid = threadIdx.x;
  float acc = 0.f;
  for (int i = tid; i < Bsz * Tsz; i += 256) acc += losses[i];
  s[tid] = acc;
  __syncthreads();
  for (int off = 128; off > 0; off >>= 1) {
    if (tid < off) s[tid] += s[tid + off];
    __syncthreads();
  }
  if (tid == 0) out[0] = s[0];
}

// ------------------------------ host driver --------------------------------

extern "C" void kernel_launch(void* const* d_in, const int* in_sizes, int n_in,
                              void* d_out, int out_size, void* d_ws,
                              size_t ws_size, hipStream_t stream) {
  const int*   word_ids = (const int*)d_in[0];
  const int*   tag_ids  = (const int*)d_in[1];
  const int*   actions  = (const int*)d_in[2];
  const float* W_word = (const float*)d_in[3];
  const float* W_tag  = (const float*)d_in[4];
  const float* A_emb  = (const float*)d_in[5];
  const float* Wih_s  = (const float*)d_in[6];
  const float* Whh_s  = (const float*)d_in[7];
  const float* bih_s  = (const float*)d_in[8];
  const float* bhh_s  = (const float*)d_in[9];
  const float* Wih_b  = (const float*)d_in[10];
  const float* Whh_b  = (const float*)d_in[11];
  const float* bih_b  = (const float*)d_in[12];
  const float* bhh_b  = (const float*)d_in[13];
  const float* Wih_a  = (const float*)d_in[14];
  const float* Whh_a  = (const float*)d_in[15];
  const float* bih_a  = (const float*)d_in[16];
  const float* bhh_a  = (const float*)d_in[17];
  const float* W_mlp  = (const float*)d_in[18];
  const float* b_mlp  = (const float*)d_in[19];
  const float* init_h = (const float*)d_in[20];
  const float* init_c = (const float*)d_in[21];
  (void)in_sizes; (void)n_in; (void)out_size; (void)ws_size;

  // ---- workspace carve-up (deterministic) ----
  size_t off = 0;
  auto take = [&](size_t bytes) -> char* {
    off = (off + 255) & ~(size_t)255;
    char* p = (char*)d_ws + off;
    off += bytes;
    return p;
  };
  bhalf* P_Wih_s = (bhalf*)take(1024ull * G4 * 2);
  bhalf* P_Whh_s = (bhalf*)take(1024ull * G4 * 2);
  bhalf* P_Wih_b = (bhalf*)take(1024ull * G4 * 2);
  bhalf* P_Whh_b = (bhalf*)take(1024ull * G4 * 2);
  bhalf* P_Wih_a = (bhalf*)take(512ull * G4 * 2);
  bhalf* P_Whh_a = (bhalf*)take(1024ull * G4 * 2);
  bhalf* Apack   = (bhalf*)take(1024ull * 1024 * 2);
  float* Xb_pre  = (float*)take(1024ull * G4 * 4);
  float* bias_s  = (float*)take(G4 * 4);
  float* bias_b  = (float*)take(G4 * 4);
  float* bias_a  = (float*)take(G4 * 4);
  float* hs_s    = (float*)take((size_t)Bsz * Dsz * H2 * 4);
  float* cs_s    = (float*)take((size_t)Bsz * Dsz * H2 * 4);
  float* hs_b    = (float*)take((size_t)Bsz * Dsz * H2 * 4);
  float* cs_b    = (float*)take((size_t)Bsz * Dsz * H2 * 4);
  bhalf* hsb_s   = (bhalf*)take((size_t)Bsz * Dsz * H2 * 2);
  bhalf* hsb_b   = (bhalf*)take((size_t)Bsz * Dsz * H2 * 2);
  float* a_h     = (float*)take(2ull * Bsz * H2 * 4);
  float* a_c     = (float*)take(2ull * Bsz * H2 * 4);
  bhalf* a_hb    = (bhalf*)take(2ull * Bsz * H2 * 2);
  float* zerovec = (float*)take(H2 * 4);
  bhalf* zerovec_bf = (bhalf*)take(H2 * 2);
  bhalf* aemb_bf = (bhalf*)take(3ull * 512 * 2);
  int*   sI      = (int*)take((size_t)Tsz * Bsz * 4);
  int*   bI      = (int*)take((size_t)Tsz * Bsz * 4);
  float* losses  = (float*)take((size_t)Bsz * Tsz * 4);

  // ---- one-time packs / precomputes ----
  pack_B_k<<<2048, 256, 0, stream>>>(Wih_s, P_Wih_s, G4, 1024);
  pack_B_k<<<2048, 256, 0, stream>>>(Whh_s, P_Whh_s, G4, 1024);
  pack_B_k<<<2048, 256, 0, stream>>>(Wih_b, P_Wih_b, G4, 1024);
  pack_B_k<<<2048, 256, 0, stream>>>(Whh_b, P_Whh_b, G4, 1024);
  pack_B_k<<<2048, 256, 0, stream>>>(Wih_a, P_Wih_a, G4, 512);
  pack_B_k<<<2048, 256, 0, stream>>>(Whh_a, P_Whh_a, G4, 1024);
  pack_x_emb_k<<<512, 256, 0, stream>>>(word_ids, tag_ids, W_word, W_tag, Apack);
  vadd_k<<<16, 256, 0, stream>>>(bih_s, bhh_s, bias_s, G4);
  vadd_k<<<16, 256, 0, stream>>>(bih_b, bhh_b, bias_b, G4);
  vadd_k<<<16, 256, 0, stream>>>(bih_a, bhh_a, bias_a, G4);
  cvt_bf_k<<<6, 256, 0, stream>>>(A_emb, aemb_bf, 3 * 512);
  init_state_k<<<4, 256, 0, stream>>>(init_h, init_c, hs_s, cs_s, hs_b, cs_b,
                                      hsb_s, hsb_b, zerovec, zerovec_bf);
  traj_k<<<1, 32, 0, stream>>>(actions, sI, bI);

  // Xb_pre = x_emb @ Wih_b^T + bias_b   (one big WMMA GEMM)
  gemm_xpre_k<<<dim3(64, 256), 32, 0, stream>>>(Apack, P_Wih_b, bias_b, Xb_pre);

  // ---- setup cells ----
  // stack guard push: p0 -> p1
  {
    UniformR r{zerovec_bf, 0, hsb_s, cs_s, SSTR, 0, hs_s, cs_s, hsb_s, SSTR, 1};
    cell_k<UniformR><<<64, 32, 0, stream>>>(r, P_Wih_s, 32, P_Whh_s, bias_s, 0);
  }
  // buffer guard push: p0 -> p1
  {
    UniformR r{zerovec_bf, 0, hsb_b, cs_b, SSTR, 0, hs_b, cs_b, hsb_b, SSTR, 1};
    cell_k<UniformR><<<64, 32, 0, stream>>>(r, P_Wih_b, 32, P_Whh_b, bias_b, 0);
  }
  // buffer word pushes: 64 sequential recurrent steps (x-part precomputed)
  for (int iw = 0; iw < Lsz; ++iw) {
    XpreR r{Xb_pre + (long long)iw * G4, (long long)Lsz * G4,
            hsb_b, cs_b, hs_b, cs_b, hsb_b, iw + 1, iw + 2};
    cell_k<XpreR><<<64, 32, 0, stream>>>(r, (const bhalf*)nullptr, 0,
                                         P_Whh_b, (const float*)nullptr, 1);
  }
  // initial shift: stack.push(buffer top at p=65): stack p1 -> p2
  {
    UniformR r{hsb_b + 65LL * H2, SSTR, hsb_s, cs_s, SSTR, 1,
               hs_s, cs_s, hsb_s, SSTR, 2};
    cell_k<UniformR><<<64, 32, 0, stream>>>(r, P_Wih_s, 32, P_Whh_s, bias_s, 0);
  }
  // initial action cell: x = A_emb[shift], h/c = init (stored at stack row 0)
  {
    UniformR r{aemb_bf, 0, hsb_s, cs_s, SSTR, 0,
               a_h, a_c, a_hb, (long long)H2, 0};
    cell_k<UniformR><<<64, 32, 0, stream>>>(r, P_Wih_a, 16, P_Whh_a, bias_a, 0);
  }

  // ---- 126 fused transition steps (one launch each) ----
  for (int t = 0; t < Tsz; ++t) {
    trans_step_k<<<193, 32, 0, stream>>>(
        t, actions, sI, bI, aemb_bf,
        P_Wih_s, P_Whh_s, bias_s,
        P_Wih_b, P_Whh_b, bias_b,
        P_Wih_a, P_Whh_a, bias_a,
        hs_s, cs_s, hs_b, cs_b, hsb_s, hsb_b,
        a_h, a_c, a_hb,
        W_mlp, b_mlp, losses);
  }

  reduce_k<<<1, 256, 0, stream>>>(losses, (float*)d_out);
}
